// BahdanauSelfAttention_24764781429505
// MI455X (gfx1250) — compile-verified, matrix operand order checked by reading, not run on hardware
//
#include <hip/hip_runtime.h>
#include <hip/hip_bf16.h>

#define BATCH 4
#define SEQ   512
#define DIM   128
#define TI    8          // i-rows per block in the scores kernel

typedef __attribute__((ext_vector_type(16))) __bf16 v16bf;
typedef __attribute__((ext_vector_type(8)))  float  v8f;

#if __has_builtin(__builtin_amdgcn_tanhf)
#define TANHF(x) __builtin_amdgcn_tanhf(x)
#else
#define TANHF(x) tanhf(x)
#endif

// CDNA5 16-bit A/B fragment K-pattern: element e of lane-half h covers
// K = {8h..8h+7} for e<8 and {16+8h..16+8h+7} for e>=8 (ISA 7.12.2).
__device__ __forceinline__ int kpat(int e, int h) {
  return ((e < 8) ? 0 : 16) + h * 8 + (e & 7);
}

// ---------------------------------------------------------------------------
// Kernel 1: Wi[b,n,e] = sum_d B[b,n,d] * W[e,d]   (B @ W^T), bf16 WMMA.
// grid = BATCH*(SEQ/16) blocks, 256 threads (8 waves); wave w -> col tile w.
// ---------------------------------------------------------------------------
__global__ void __launch_bounds__(256)
wi_gemm_kernel(const float* __restrict__ B, const float* __restrict__ W,
               float* __restrict__ Wi) {
  __shared__ __bf16 sA[16][DIM + 8];    // 16 rows of B (M x K)
  __shared__ __bf16 sW[DIM][DIM + 8];   // full W   (e x d) == B-matrix^T source

  const int tid = threadIdx.x;
  const int b   = blockIdx.x / (SEQ / 16);
  const int rt  = blockIdx.x % (SEQ / 16);

  const float* Brow = B + ((size_t)b * SEQ + rt * 16) * DIM;
  for (int i = tid; i < 16 * DIM; i += 256) {
    int r = i >> 7, c = i & 127;
    sA[r][c] = (__bf16)Brow[r * DIM + c];
  }
  for (int i = tid; i < DIM * DIM; i += 256) {
    int r = i >> 7, c = i & 127;
    sW[r][c] = (__bf16)W[i];
  }
  __syncthreads();

  const int wave = tid >> 5;            // col tile 0..7 (DIM/16 == 8)
  const int lane = tid & 31;
  const int h    = lane >> 4;
  const int mn   = lane & 15;

  v8f acc = {};
#pragma unroll
  for (int k0 = 0; k0 < DIM; k0 += 32) {
    v16bf a, w;
#pragma unroll
    for (int e = 0; e < 16; ++e) {
      const int k = k0 + kpat(e, h);
      a[e] = sA[mn][k];                 // A(m=mn, k)
      w[e] = sW[wave * 16 + mn][k];     // B(k, n=mn) = W[n_global][k]
    }
    acc = __builtin_amdgcn_wmma_f32_16x16x32_bf16(false, a, false, w,
                                                  (short)0, acc, false, false);
  }

  float* out = Wi + ((size_t)b * SEQ + rt * 16) * DIM + wave * 16 + mn;
#pragma unroll
  for (int r = 0; r < 8; ++r)
    out[(size_t)(r + 8 * h) * DIM] = acc[r];   // C layout: row = r + 8h, col = mn
}

// ---------------------------------------------------------------------------
// Kernel 2: scores[i,j] = v . tanh(Wi[i] + Wi[j]); softmax over j; attn -> bf16.
// One block per (b, 8 i-rows): each global Wi[j,:] row is reused across the 8
// register accumulators (8x L2-traffic cut -> tanh-rate bound).
// 256 threads, each handles j = tid and tid+256.
// ---------------------------------------------------------------------------
__global__ void __launch_bounds__(256)
scores_softmax_kernel(const float* __restrict__ Wi, const float* __restrict__ v,
                      __bf16* __restrict__ attn) {
  __shared__ float wii[TI][DIM];        // 8 query rows
  __shared__ float vv[DIM];
  __shared__ float sc[TI][SEQ];         // raw scores for the 8 rows
  __shared__ float red[256];

  const int tid = threadIdx.x;
  const int b   = blockIdx.x / (SEQ / TI);
  const int it  = blockIdx.x % (SEQ / TI);
  const int i0  = it * TI;
  const float* wiB = Wi + (size_t)b * SEQ * DIM;

  for (int idx = tid; idx < TI * DIM; idx += 256)
    wii[idx >> 7][idx & 127] = wiB[(size_t)(i0 + (idx >> 7)) * DIM + (idx & 127)];
  if (tid < DIM) vv[tid] = v[tid];
  __syncthreads();

#pragma unroll
  for (int jj = 0; jj < 2; ++jj) {
    const int j = tid + jj * 256;
    const float4* wj = (const float4*)(wiB + (size_t)j * DIM);
    __builtin_prefetch(wiB + (size_t)(tid + (jj + 1) * 256) * DIM, 0, 1);

    float s[TI];
#pragma unroll
    for (int i = 0; i < TI; ++i) s[i] = 0.f;

#pragma unroll 2
    for (int d4 = 0; d4 < DIM / 4; ++d4) {
      const float4 w4 = wj[d4];
      const float4 v4 = ((const float4*)vv)[d4];
#pragma unroll
      for (int i = 0; i < TI; ++i) {
        const float4 wi4 = ((const float4*)wii[i])[d4];
        s[i] += v4.x * TANHF(wi4.x + w4.x);
        s[i] += v4.y * TANHF(wi4.y + w4.y);
        s[i] += v4.z * TANHF(wi4.z + w4.z);
        s[i] += v4.w * TANHF(wi4.w + w4.w);
      }
    }
#pragma unroll
    for (int i = 0; i < TI; ++i) sc[i][j] = s[i];
  }
  __syncthreads();

  // softmax for each of the 8 rows (amortized over the block)
  for (int r = 0; r < TI; ++r) {
    red[tid] = fmaxf(sc[r][tid], sc[r][tid + 256]);
    __syncthreads();
    for (int o = 128; o > 0; o >>= 1) {
      if (tid < o) red[tid] = fmaxf(red[tid], red[tid + o]);
      __syncthreads();
    }
    const float M = red[0];
    __syncthreads();

    const float e0 = __expf(sc[r][tid] - M);
    const float e1 = __expf(sc[r][tid + 256] - M);
    red[tid] = e0 + e1;
    __syncthreads();
    for (int o = 128; o > 0; o >>= 1) {
      if (tid < o) red[tid] += red[tid + o];
      __syncthreads();
    }
    const float inv = 1.f / red[0];

    __bf16* arow = attn + ((size_t)b * SEQ + i0 + r) * SEQ;
    arow[tid]       = (__bf16)(e0 * inv);
    arow[tid + 256] = (__bf16)(e1 * inv);
    __syncthreads();   // red[] reused next row
  }
}

// ---------------------------------------------------------------------------
// Kernel 3: C[b] = attn[b] (512x512 bf16) @ B[b] (512x128 -> bf16), f32 acc.
// K staged 64 wide -> 2 WMMAs per barrier round.
// grid = BATCH*(SEQ/16), 256 threads (8 waves); wave w -> col tile w.
// ---------------------------------------------------------------------------
__global__ void __launch_bounds__(256)
av_gemm_kernel(const __bf16* __restrict__ attn, const float* __restrict__ B,
               float* __restrict__ C) {
  __shared__ __bf16 sA[16][64 + 8];     // attn tile (M x 64)
  __shared__ __bf16 sB[64][DIM + 8];    // B tile    (64 x N)

  const int tid = threadIdx.x;
  const int b   = blockIdx.x / (SEQ / 16);
  const int rt  = blockIdx.x % (SEQ / 16);

  const __bf16* aRow = attn + ((size_t)b * SEQ + rt * 16) * SEQ;
  const float*  bB   = B + (size_t)b * SEQ * DIM;

  const int wave = tid >> 5;
  const int lane = tid & 31;
  const int h    = lane >> 4;
  const int mn   = lane & 15;

  v8f acc = {};
  for (int k0 = 0; k0 < SEQ; k0 += 64) {
    // stage attn 16x64 and B 64x128 (f32 -> bf16)
    for (int i = tid; i < 16 * 64; i += 256) {
      int r = i >> 6, c = i & 63;
      sA[r][c] = aRow[(size_t)r * SEQ + k0 + c];
    }
    for (int i = tid; i < 64 * DIM; i += 256) {
      int r = i >> 7, c = i & 127;
      sB[r][c] = (__bf16)bB[(size_t)(k0 + r) * DIM + c];
    }
    if (k0 + 64 < SEQ)
      __builtin_prefetch(bB + (size_t)(k0 + 64 + (tid >> 3)) * DIM + (tid & 7) * 16, 0, 1);
    __syncthreads();

#pragma unroll
    for (int ks = 0; ks < 64; ks += 32) {
      v16bf a, bb;
#pragma unroll
      for (int e = 0; e < 16; ++e) {
        const int k = ks + kpat(e, h);
        a[e]  = sA[mn][k];
        bb[e] = sB[k][wave * 16 + mn];
      }
      acc = __builtin_amdgcn_wmma_f32_16x16x32_bf16(false, a, false, bb,
                                                    (short)0, acc, false, false);
    }
    __syncthreads();
  }

  float* out = C + ((size_t)b * SEQ + rt * 16) * DIM + wave * 16 + mn;
#pragma unroll
  for (int r = 0; r < 8; ++r)
    out[(size_t)(r + 8 * h) * DIM] = acc[r];
}

// ---------------------------------------------------------------------------
extern "C" void kernel_launch(void* const* d_in, const int* in_sizes, int n_in,
                              void* d_out, int out_size, void* d_ws, size_t ws_size,
                              hipStream_t stream) {
  (void)in_sizes; (void)n_in; (void)out_size; (void)ws_size;

  const float* B = (const float*)d_in[0];
  const float* W = (const float*)d_in[1];
  const float* v = (const float*)d_in[2];
  float* C = (float*)d_out;

  // workspace layout: Wi f32 (4*512*128*4 = 1 MB) | attn bf16 (4*512*512*2 = 2 MB)
  float*  Wi   = (float*)d_ws;
  __bf16* attn = (__bf16*)((char*)d_ws + (size_t)BATCH * SEQ * DIM * sizeof(float));

  wi_gemm_kernel<<<BATCH * (SEQ / 16), 256, 0, stream>>>(B, W, Wi);
  scores_softmax_kernel<<<BATCH * (SEQ / TI), 256, 0, stream>>>(Wi, v, attn);
  av_gemm_kernel<<<BATCH * (SEQ / 16), 256, 0, stream>>>(attn, B, C);
}
